// ConvTransformerEncoderLayer_28020366639677
// MI455X (gfx1250) — matrix operand order repaired and checked by reading, not compiled, not run on hardware
//
#include <hip/hip_runtime.h>
#include <math.h>

// ---------------- model dims ----------------
constexpr int kL   = 256;   // sequence length
constexpr int kD   = 256;   // d_model
constexpr int kNH  = 8;     // MHA heads
constexpr int kDK  = 32;    // MHA head dim
constexpr int kFF  = 1024;  // FFN hidden
constexpr int kP   = 64;    // pair dim
constexpr int kTAH = 4;     // tri-attn heads
constexpr int kTAD = 16;    // tri-attn head dim
constexpr int kDM  = 32;    // outer-product-mean dim

// ------- d_in index map (jax pytree flatten: dict keys sorted, DFS) -------
enum {
  IN_SRC = 0, IN_PAIR = 1, IN_MASK = 2,
  P_CONV_B = 3, P_CONV_W = 4,
  P_LIN1_B = 5, P_LIN1_W = 6, P_LIN2_B = 7, P_LIN2_W = 8,
  P_MHA_FC = 9, P_MHA_LN_B = 10, P_MHA_LN_G = 11, P_MHA_WK = 12, P_MHA_WQ = 13, P_MHA_WV = 14,
  P_NORM1_B = 15, P_NORM1_G = 16, P_NORM2_B = 17, P_NORM2_G = 18, P_NORM3_B = 19, P_NORM3_G = 20,
  P_OPM1_B = 21, P_OPM1_W = 22, P_OPM2_B = 23, P_OPM2_W = 24,
  P_P2H_W = 25,
  P_PT_B1 = 26, P_PT_B2 = 27, P_PT_LN_B = 28, P_PT_LN_G = 29, P_PT_W1 = 30, P_PT_W2 = 31,
  P_PWN_B = 32, P_PWN_G = 33,
  TAI_BASE = 34, TAO_BASE = 42, TMI_BASE = 50, TMO_BASE = 66
};
// tri-attn sub-dict (sorted): gate_b, gate_w, norm_b, norm_g, out_b, out_w, pair_w, qkv_w
enum { TA_GATE_B = 0, TA_GATE_W, TA_NORM_B, TA_NORM_G, TA_OUT_B, TA_OUT_W, TA_PAIR_W, TA_QKV_W };
// tri-mult sub-dict (sorted)
enum { TM_LGATE_B = 0, TM_LGATE_W, TM_LPROJ_B, TM_LPROJ_W, TM_NORM_B, TM_NORM_G,
       TM_OGATE_B, TM_OGATE_W, TM_ONORM_B, TM_ONORM_G, TM_OUT_B, TM_OUT_W,
       TM_RGATE_B, TM_RGATE_W, TM_RPROJ_B, TM_RPROJ_W };

#define PF(i) ((const float*)d_in[(i)])

// ---------------- WMMA types ----------------
typedef __attribute__((ext_vector_type(16))) __bf16 v16bf;
typedef __attribute__((ext_vector_type(8)))  __bf16 v8bf;
typedef __attribute__((ext_vector_type(8)))  float  v8f;

__device__ __forceinline__ void pack8(v16bf& f, int o, const float4& x, const float4& y) {
  f[o + 0] = (__bf16)x.x; f[o + 1] = (__bf16)x.y; f[o + 2] = (__bf16)x.z; f[o + 3] = (__bf16)x.w;
  f[o + 4] = (__bf16)y.x; f[o + 5] = (__bf16)y.y; f[o + 6] = (__bf16)y.z; f[o + 7] = (__bf16)y.w;
}

// Generic batched, strided GEMM (k-contiguous operands, K%32==0 REQUIRED):
//   C[z][m,n] (+)= alpha * sum_k A[z][m,k] * B[z][n,k]  (+ bias[n])
// Per-operand dtype: abf/bbf = 1 -> operand stored bf16; bf16 fragments load
// directly with zero cvt in the loop.  C is always fp32.
// z = blockIdx.z decomposed as (zo = z / binner, zi = z % binner).
// One wave computes a 16 x (16*ntiles) C block (ntiles<=4): A fragment reused
// across 4 WMMAs per K-step; b128 clauses + prefetch of the next K-step.
struct GemmDesc {
  const void* A; const void* B; float* C; const float* bias;
  int M, N, K;
  long long am, abase, abo, abi;
  long long bn, bbase, bbo, bbi;
  long long cm, cn, cbase, cbo, cbi;
  int binner; float alpha; int accum; int abf; int bbf;
};

__global__ __launch_bounds__(32) void k_wmma_gemm(GemmDesc g) {
  const int lane = threadIdx.x;
  const int zo = (int)blockIdx.z / g.binner;
  const int zi = (int)blockIdx.z % g.binner;
  const int tm  = blockIdx.x * 16;
  const int tn0 = blockIdx.y * 64;
  const int ntiles = min(4, (g.N - tn0 + 15) >> 4);   // wave-uniform
  // ISA fragment layouts (CDNA5 ISA §7.12.2, wave32):
  //  A (16x32 bf16): m = lane&15, k = 8*(lane>>4) + 16*(e>>3) + (e&7)
  //  B (32x16 bf16): n = lane&15, k = 8*(lane>>4) + 16*(e>>3) + (e&7)
  //  C (16x16 f32):  m = v + 8*(lane>>4), n = lane&15
  const int mrow = tm + (lane & 15);
  const int kba = (lane >> 4) * 8;
  const int kbb = (lane >> 4) * 8;
  // Clamp rows: OOB m/n fragments only feed C rows/cols that are never stored.
  const int mc = mrow < g.M ? mrow : g.M - 1;
  const long long aoff = g.abase + (long long)zo * g.abo + (long long)zi * g.abi + (long long)mc * g.am;
  long long boff[4];
#pragma unroll
  for (int t = 0; t < 4; ++t) {
    int ncol = tn0 + t * 16 + (lane & 15);
    int nc = ncol < g.N ? ncol : g.N - 1;
    boff[t] = g.bbase + (long long)zo * g.bbo + (long long)zi * g.bbi + (long long)nc * g.bn;
  }
  float* C = g.C + g.cbase + (long long)zo * g.cbo + (long long)zi * g.cbi;
  const v8f vzero = {0.f, 0.f, 0.f, 0.f, 0.f, 0.f, 0.f, 0.f};
  v8f acc[4];
#pragma unroll
  for (int t = 0; t < 4; ++t) acc[t] = vzero;

  for (int k0 = 0; k0 < g.K; k0 += 32) {
    v16bf fa;
    if (g.abf) {
      const __bf16* ap = (const __bf16*)g.A + aoff + k0 + kba;
      v8bf lo = *(const v8bf*)(ap);         // k = kba..kba+7   (16 B)
      v8bf hi = *(const v8bf*)(ap + 16);    // k = kba+16..+23  (16 B)
      fa = __builtin_shufflevector(lo, hi, 0, 1, 2, 3, 4, 5, 6, 7,
                                   8, 9, 10, 11, 12, 13, 14, 15);
      if (k0 + 32 < g.K) __builtin_prefetch(ap + 32, 0, 0);
    } else {
      const float* ap = (const float*)g.A + aoff + k0 + kba;
      float4 a0 = *(const float4*)(ap);
      float4 a1 = *(const float4*)(ap + 4);
      float4 a2 = *(const float4*)(ap + 16);
      float4 a3 = *(const float4*)(ap + 20);
      pack8(fa, 0, a0, a1); pack8(fa, 8, a2, a3);
      if (k0 + 32 < g.K) __builtin_prefetch(ap + 32, 0, 0);   // global_prefetch_b8
    }
#pragma unroll
    for (int t = 0; t < 4; ++t) {
      if (t < ntiles) {
        v16bf fb;
        if (g.bbf) {
          const __bf16* bp = (const __bf16*)g.B + boff[t] + k0 + kbb;
          v8bf blo = *(const v8bf*)(bp);        // k = kbb..kbb+7
          v8bf bhi = *(const v8bf*)(bp + 16);   // k = kbb+16..+23
          fb = __builtin_shufflevector(blo, bhi, 0, 1, 2, 3, 4, 5, 6, 7,
                                       8, 9, 10, 11, 12, 13, 14, 15);
          if (k0 + 32 < g.K) __builtin_prefetch(bp + 32, 0, 0);
        } else {
          const float* bp = (const float*)g.B + boff[t] + k0 + kbb;
          float4 b0 = *(const float4*)(bp);
          float4 b1 = *(const float4*)(bp + 4);
          float4 b2 = *(const float4*)(bp + 16);
          float4 b3 = *(const float4*)(bp + 20);
          pack8(fb, 0, b0, b1); pack8(fb, 8, b2, b3);
          if (k0 + 32 < g.K) __builtin_prefetch(bp + 32, 0, 0);
        }
        acc[t] = __builtin_amdgcn_wmma_f32_16x16x32_bf16(false, fa, false, fb, (short)0, acc[t], false, false);
      }
    }
  }
#pragma unroll
  for (int t = 0; t < 4; ++t) {
    if (t < ntiles) {
#pragma unroll
      for (int r = 0; r < 8; ++r) {
        int m = tm + r + ((lane >> 4) << 3);
        int n = tn0 + t * 16 + (lane & 15);
        if (m < g.M && n < g.N) {
          long long idx = (long long)m * g.cm + (long long)n * g.cn;
          float val = acc[t][r] * g.alpha;
          if (g.bias) val += g.bias[n];
          if (g.accum) val += C[idx];
          C[idx] = val;
        }
      }
    }
  }
}

// ---------------- LayerNorm (one wave per row, optional residual, f32 or bf16 out) ----------------
__global__ __launch_bounds__(32) void k_layernorm(const float* __restrict__ in,
                                                  const float* __restrict__ res,
                                                  void* __restrict__ out, int obf,
                                                  const float* __restrict__ g,
                                                  const float* __restrict__ b,
                                                  int width, float eps) {
  long long row = blockIdx.x;
  const float* x = in + row * width;
  const float* r = res ? res + row * width : nullptr;
  int lane = threadIdx.x;
  float s = 0.f, ss = 0.f;
  for (int c = lane; c < width; c += 32) {
    float v = x[c] + (r ? r[c] : 0.f);
    s += v; ss += v * v;
  }
  for (int off = 16; off; off >>= 1) {
    s += __shfl_xor(s, off, 32);
    ss += __shfl_xor(ss, off, 32);
  }
  float mu = s / width;
  float var = ss / width - mu * mu;
  float inv = rsqrtf(var + eps);
  for (int c = lane; c < width; c += 32) {
    float v = x[c] + (r ? r[c] : 0.f);
    float o = (v - mu) * inv * g[c] + b[c];
    if (obf) ((__bf16*)out)[row * width + c] = (__bf16)o;
    else     ((float*)out)[row * width + c] = o;
  }
}

// ---------------- masked softmax over last dim (len kL), one wave per row ----------------
// Reads fp32 logits [z][i][j]; writes bf16 probabilities to bout (same layout) —
// bout feeds the AV GEMM A operand directly.  h = z % nh; bias indexed
// [(i*kL+j)*nh + h].  mode 0 (MHA): mask where i!=j && m[i]*m[j]==0;
// mode 1 (tri): mask where m[i]!=m[j].  Masked logits become -1e-9 (faithful).
__global__ __launch_bounds__(32) void k_softmax(const float* __restrict__ logits,
                                                __bf16* __restrict__ bout,
                                                const float* __restrict__ bias,
                                                int nh, const int* __restrict__ mask,
                                                int mode) {
  int i = blockIdx.x;
  int z = blockIdx.y;
  int h = z % nh;
  const float* row = logits + ((long long)z * kL + i) * kL;
  __bf16* orow = bout + ((long long)z * kL + i) * kL;
  int lane = threadIdx.x;
  int mi = mask[i];
  float vals[kL / 32];
  float mx = -3.4e38f;
#pragma unroll
  for (int t = 0; t < kL / 32; ++t) {
    int j = lane + t * 32;
    float v = row[j];
    if (bias) v += bias[((long long)i * kL + j) * nh + h];
    int mj = mask[j];
    bool masked = (mode == 0) ? ((i != j) && (mi * mj == 0)) : (mi != mj);
    if (masked) v = -1e-9f;
    vals[t] = v;
    mx = fmaxf(mx, v);
  }
  for (int off = 16; off; off >>= 1) mx = fmaxf(mx, __shfl_xor(mx, off, 32));
  float sum = 0.f;
#pragma unroll
  for (int t = 0; t < kL / 32; ++t) { vals[t] = __expf(vals[t] - mx); sum += vals[t]; }
  for (int off = 16; off; off >>= 1) sum += __shfl_xor(sum, off, 32);
  float inv = 1.f / sum;
#pragma unroll
  for (int t = 0; t < kL / 32; ++t) orow[lane + t * 32] = (__bf16)(vals[t] * inv);
}

// ---------------- layout transforms (emit bf16 GEMM operand streams) ----------------
// Pair channel transpose for the triangle einsum:
// out[(d*kL + a)*kL + b] = in[(a*kL+b)*kP + d]  (swap=0)  or  in[(b*kL+a)*kP + d]  (swap=1)
__global__ void k_dtrans(const float* __restrict__ in, __bf16* __restrict__ out, int swap) {
  int idx = blockIdx.x * 256 + threadIdx.x;
  if (idx >= kL * kL * kP) return;
  int b = idx & (kL - 1);
  int a = (idx >> 8) & (kL - 1);
  int d = idx >> 16;
  int ab = swap ? (b * kL + a) : (a * kL + b);
  out[idx] = (__bf16)in[((long long)ab << 6) + d];
}

// Generic V transpose: out[(z*dn + d)*kL + j] = in[base + zo*zos + zi*zis + j*js + d]
__global__ void k_vtrans(const float* __restrict__ in, __bf16* __restrict__ out,
                         long long base, long long js, long long zos, long long zis,
                         int binner, int dn, int nz) {
  int idx = blockIdx.x * 256 + threadIdx.x;
  if (idx >= nz * dn * kL) return;
  int j = idx & (kL - 1);
  int d = (idx >> 8) % dn;
  int z = idx / (kL * dn);
  int zo = z / binner, zi = z % binner;
  out[idx] = (__bf16)in[base + (long long)zo * zos + (long long)zi * zis + (long long)j * js + d];
}

// Tri-attn Q/K repack: out[(z*kL + i)*32 + d] = d<16 ? in[base+zo*zos+zi*zis+i*is+d] : 0
// Zero-pads K=16 head dim to 32 so the logits GEMM rides the bf16 WMMA fast path.
__global__ void k_qtrans(const float* __restrict__ in, __bf16* __restrict__ out,
                         long long base, long long is, long long zos, long long zis,
                         int binner, int nz) {
  int idx = blockIdx.x * 256 + threadIdx.x;
  if (idx >= nz * kL * 32) return;
  int d = idx & 31;
  int i = (idx >> 5) & (kL - 1);
  int z = idx >> 13;
  int zo = z / binner, zi = z % binner;
  float v = (d < kTAD)
      ? in[base + (long long)zo * zos + (long long)zi * zis + (long long)i * is + d]
      : 0.f;
  out[idx] = (__bf16)v;
}

// ---------------- small elementwise kernels ----------------
__global__ void k_maskmul(const float* __restrict__ src, const int* __restrict__ mask,
                          float* __restrict__ out, int n) {
  int idx = blockIdx.x * 256 + threadIdx.x;
  if (idx >= n) return;
  out[idx] = src[idx] * (float)mask[idx >> 8];   // kD == 256
}

// length-conv, kernel 3, padding 1, plus residual + bias. srcm (L,D); conv_w (O,I,3).
__global__ void k_conv(const float* __restrict__ srcm, const float* __restrict__ w,
                       const float* __restrict__ b, float* __restrict__ out) {
  int idx = blockIdx.x * 256 + threadIdx.x;   // 65536 threads
  int c = idx & (kD - 1);
  int i = idx >> 8;
  float acc = srcm[idx] + b[c];
  for (int dk = 0; dk < 3; ++dk) {
    int ii = i + dk - 1;
    if (ii < 0 || ii >= kL) continue;
    const float* wrow = w + (long long)c * (kD * 3) + dk;
    const float* s = srcm + (long long)ii * kD;
    for (int ci = 0; ci < kD; ++ci) acc += wrow[ci * 3] * s[ci];
  }
  out[idx] = acc;
}

__global__ void k_sigmoid(float* __restrict__ a, int n) {
  int idx = blockIdx.x * 256 + threadIdx.x;
  if (idx >= n) return;
  a[idx] = 1.f / (1.f + __expf(-a[idx]));
}
__global__ void k_mul(float* __restrict__ a, const float* __restrict__ b, int n) {
  int idx = blockIdx.x * 256 + threadIdx.x;
  if (idx >= n) return;
  a[idx] *= b[idx];
}
__global__ void k_relu(float* __restrict__ a, int n) {
  int idx = blockIdx.x * 256 + threadIdx.x;
  if (idx >= n) return;
  a[idx] = fmaxf(a[idx], 0.f);
}
__global__ void k_gelu(float* __restrict__ a, int n) {
  int idx = blockIdx.x * 256 + threadIdx.x;
  if (idx >= n) return;
  float x = a[idx];
  a[idx] = 0.5f * x * (1.f + erff(x * 0.70710678118654752f));   // exact gelu
}
// buf[(i*kL+j)*kP + d] *= m[i]*m[j] * sigmoid(gate[same])
__global__ void k_gatemask(float* __restrict__ buf, const float* __restrict__ gate,
                           const int* __restrict__ mask, int n) {
  int idx = blockIdx.x * 256 + threadIdx.x;
  if (idx >= n) return;
  int i = idx >> 14;           // kL*kP == 16384
  int j = (idx >> 6) & (kL - 1);
  float mm = (float)(mask[i] * mask[j]);
  buf[idx] = buf[idx] * mm * (1.f / (1.f + __expf(-gate[idx])));
}

// ---------------- host helpers ----------------
static GemmDesc gd(const void* A, const void* B, float* C, const float* bias,
                   int M, int N, int K) {
  GemmDesc g;
  g.A = A; g.B = B; g.C = C; g.bias = bias; g.M = M; g.N = N; g.K = K;
  g.am = K; g.abase = 0; g.abo = 0; g.abi = 0;
  g.bn = K; g.bbase = 0; g.bbo = 0; g.bbi = 0;
  g.cm = N; g.cn = 1; g.cbase = 0; g.cbo = 0; g.cbi = 0;
  g.binner = 1; g.alpha = 1.f; g.accum = 0; g.abf = 0; g.bbf = 0;
  return g;
}
static inline void rungemm(hipStream_t st, const GemmDesc& g, int batch = 1) {
  dim3 grid((unsigned)((g.M + 15) / 16), (unsigned)((g.N + 63) / 64), (unsigned)batch);
  k_wmma_gemm<<<grid, 32, 0, st>>>(g);
}
static inline int ewb(int n) { return (n + 255) / 256; }

// x + tri_mult(x): triangle-multiplicative update, accumulated into PAIR.
static void tri_mult_block(hipStream_t st, void* const* d_in, int base,
                           float* PAIR, __bf16* XNB, float* B0, float* B1, float* B2,
                           float* B3, float* B4, const int* mask, bool outgoing) {
  const int NPAIR = kL * kL * kP;
  k_layernorm<<<kL * kL, 32, 0, st>>>(PAIR, nullptr, XNB, 1, PF(base + TM_NORM_G), PF(base + TM_NORM_B), kP, 1e-5f);
  GemmDesc g = gd(XNB, PF(base + TM_LPROJ_W), B0, PF(base + TM_LPROJ_B), kL * kL, kP, kP);
  g.abf = 1; rungemm(st, g);
  g = gd(XNB, PF(base + TM_LGATE_W), B3, PF(base + TM_LGATE_B), kL * kL, kP, kP);
  g.abf = 1; rungemm(st, g);
  k_gatemask<<<ewb(NPAIR), 256, 0, st>>>(B0, B3, mask, NPAIR);               // left
  g = gd(XNB, PF(base + TM_RPROJ_W), B1, PF(base + TM_RPROJ_B), kL * kL, kP, kP);
  g.abf = 1; rungemm(st, g);
  g = gd(XNB, PF(base + TM_RGATE_W), B3, PF(base + TM_RGATE_B), kL * kL, kP, kP);
  g.abf = 1; rungemm(st, g);
  k_gatemask<<<ewb(NPAIR), 256, 0, st>>>(B1, B3, mask, NPAIR);               // right
  g = gd(XNB, PF(base + TM_OGATE_W), B2, PF(base + TM_OGATE_B), kL * kL, kP, kP);
  g.abf = 1; rungemm(st, g);
  k_sigmoid<<<ewb(NPAIR), 256, 0, st>>>(B2, NPAIR);                          // og
  // Transpose to bf16 [d][row][k] (k-contiguous): the 64x(256^3) einsum then
  // streams pure b128 bf16 with no cvt.  XNB is dead here; reuse for A.
  //  outgoing: A[i,k]=left[i,k,d] (swap 0), B[j,k]=right[j,k,d] (swap 0)
  //  incoming: A[i,k]=right[k,i,d] (swap 1), B[j,k]=left[k,j,d] (swap 1)
  __bf16* TA = XNB;
  __bf16* TB = (__bf16*)B4;
  k_dtrans<<<ewb(NPAIR), 256, 0, st>>>(outgoing ? B0 : B1, TA, outgoing ? 0 : 1);
  k_dtrans<<<ewb(NPAIR), 256, 0, st>>>(outgoing ? B1 : B0, TB, outgoing ? 0 : 1);
  GemmDesc e = gd(TA, TB, B3, nullptr, kL, kL, kL);
  e.abi = (long long)kL * kL;
  e.bbi = (long long)kL * kL;
  e.cm = (long long)kL * kP; e.cn = kP; e.cbi = 1;
  e.binner = kP; e.abf = 1; e.bbf = 1;
  rungemm(st, e, kP);
  k_layernorm<<<kL * kL, 32, 0, st>>>(B3, nullptr, B3, 0, PF(base + TM_ONORM_G), PF(base + TM_ONORM_B), kP, 1e-5f);
  k_mul<<<ewb(NPAIR), 256, 0, st>>>(B3, B2, NPAIR);
  g = gd(B3, PF(base + TM_OUT_W), PAIR, PF(base + TM_OUT_B), kL * kL, kP, kP);
  g.accum = 1;
  rungemm(st, g);
}

// x + tri_attn(x): triangle attention (row- or col-wise), accumulated into PAIR.
static void tri_attn_block(hipStream_t st, void* const* d_in, int base,
                           float* PAIR, __bf16* XNB, float* GATE, float* AOUT,
                           float* QKV, float* BPB, float* LOG, __bf16* LOGB,
                           __bf16* QT, __bf16* KT, __bf16* VT,
                           const int* mask, bool rowwise) {
  const int NPAIR = kL * kL * kP;
  const long long QLD = 3 * kTAH * kTAD;   // 192: qkv row stride
  k_layernorm<<<kL * kL, 32, 0, st>>>(PAIR, nullptr, XNB, 1, PF(base + TA_NORM_G), PF(base + TA_NORM_B), kP, 1e-5f);
  GemmDesc g = gd(XNB, PF(base + TA_QKV_W), QKV, nullptr, kL * kL, (int)QLD, kP);
  g.abf = 1; rungemm(st, g);
  g = gd(XNB, PF(base + TA_PAIR_W), BPB, nullptr, kL * kL, kTAH, kP);
  g.abf = 1; rungemm(st, g);
  g = gd(XNB, PF(base + TA_GATE_W), GATE, PF(base + TA_GATE_B), kL * kL, kP, kP);
  g.abf = 1; rungemm(st, g);
  k_sigmoid<<<ewb(NPAIR), 256, 0, st>>>(GATE, NPAIR);
  // Q/K -> bf16 [z=(r|l,h)][i][32] (zero-padded K) and V -> bf16 [z][d][j].
  const long long IS  = rowwise ? QLD : (long long)kL * QLD;   // i stride in qkv
  const long long ZOS = rowwise ? (long long)kL * QLD : QLD;   // (r|l) stride
  const int NZ = kL * kTAH;
  k_qtrans<<<ewb(NZ * kL * 32), 256, 0, st>>>(QKV, QT, 0,   IS, ZOS, 16, kTAH, NZ);
  k_qtrans<<<ewb(NZ * kL * 32), 256, 0, st>>>(QKV, KT, 64,  IS, ZOS, 16, kTAH, NZ);
  k_vtrans<<<ewb(NZ * kTAD * kL), 256, 0, st>>>(QKV, VT, 128, IS, ZOS, 16, kTAH, kTAD, NZ);
  const int CB = 32;   // rows/cols per chunk -> 33.5 MB logits scratch (fits L2)
  for (int c0 = 0; c0 < kL; c0 += CB) {
    GemmDesc lg = gd(QT, KT, LOG, nullptr, kL, kL, 32);        // bf16/bf16 fast path
    lg.abase = (long long)c0 * kTAH * kL * 32;
    lg.am = 32; lg.abo = (long long)kTAH * kL * 32; lg.abi = (long long)kL * 32;
    lg.bbase = (long long)c0 * kTAH * kL * 32;
    lg.bn = 32; lg.bbo = (long long)kTAH * kL * 32; lg.bbi = (long long)kL * 32;
    lg.cm = kL; lg.cn = 1; lg.cbo = (long long)kTAH * kL * kL; lg.cbi = (long long)kL * kL;
    lg.binner = kTAH; lg.alpha = 0.25f;        // 1/sqrt(TA_DIM)
    lg.abf = 1; lg.bbf = 1;
    rungemm(st, lg, CB * kTAH);
    k_softmax<<<dim3(kL, CB * kTAH), 32, 0, st>>>(LOG, LOGB, BPB, kTAH, mask, 1);
    GemmDesc og = gd(LOGB, VT, AOUT, nullptr, kL, kTAD, kL);   // bf16/bf16 fast path
    og.abo = (long long)kTAH * kL * kL; og.abi = (long long)kL * kL;
    og.bbase = (long long)c0 * kTAH * kTAD * kL;
    og.bn = kL;
    og.bbo = (long long)kTAH * kTAD * kL; og.bbi = (long long)kTAD * kL;
    if (rowwise) {
      og.cbase = (long long)c0 * kL * kP;  og.cm = kP;                og.cn = 1;
      og.cbo = (long long)kL * kP;         og.cbi = kTAD;
    } else {
      og.cbase = (long long)c0 * kP;       og.cm = (long long)kL * kP; og.cn = 1;
      og.cbo = kP;                         og.cbi = kTAD;
    }
    og.binner = kTAH; og.abf = 1; og.bbf = 1;
    rungemm(st, og, CB * kTAH);
  }
  k_mul<<<ewb(NPAIR), 256, 0, st>>>(AOUT, GATE, NPAIR);
  g = gd(AOUT, PF(base + TA_OUT_W), PAIR, PF(base + TA_OUT_B), kL * kL, kP, kP);
  g.accum = 1;
  rungemm(st, g);
}

// ---------------- entry point ----------------
extern "C" void kernel_launch(void* const* d_in, const int* in_sizes, int n_in,
                              void* d_out, int out_size, void* d_ws, size_t ws_size,
                              hipStream_t stream) {
  (void)in_sizes; (void)n_in; (void)out_size; (void)ws_size;
  hipStream_t st = stream;
  const float* src  = PF(IN_SRC);
  const int*   mask = (const int*)d_in[IN_MASK];

  // ---- workspace (bump allocator, ~220 MB total) ----
  char* wp = (char*)d_ws;
  auto alloc = [&](size_t bytes) -> void* {
    void* p = (void*)wp;
    wp += (bytes + 255) & ~(size_t)255;
    return p;
  };
  const size_t SRC_B  = (size_t)kL * kD * 4;           // 256 KB
  const size_t PAIR_B = (size_t)kL * kL * kP * 4;      // 16.78 MB
  float* S0  = (float*)alloc(SRC_B);   // masked src
  float* S1  = (float*)alloc(SRC_B);   // conv out
  float* S2  = (float*)alloc(SRC_B);   // norm3(x)
  float* S3  = (float*)alloc(SRC_B);
  float* S4  = (float*)alloc(SRC_B);
  float* S5  = (float*)alloc(SRC_B);
  float* S6  = (float*)alloc(SRC_B);   // final src
  __bf16* VTM = (__bf16*)alloc(SRC_B / 2);             // MHA V bf16 [h][d][j]
  float* FFB = (float*)alloc((size_t)kL * kFF * 4);
  float* ATT = (float*)alloc((size_t)kNH * kL * kL * 4);
  float* PWB = (float*)alloc((size_t)kL * kL * kNH * 4);
  float* SVEC = (float*)alloc((size_t)kL * kDM * 4);
  float* TBUF = (float*)alloc((size_t)kL * kP * kDM * 4);
  float* PAIR = (float*)alloc(PAIR_B);
  __bf16* XNB = (__bf16*)alloc(PAIR_B / 2);            // pair LN out (bf16, GEMM A)
  float* B0   = (float*)alloc(PAIR_B);
  float* B1   = (float*)alloc(PAIR_B);
  float* B2   = (float*)alloc(PAIR_B);   // free in tri-attn -> QT
  float* B3   = (float*)alloc(PAIR_B);   // free in tri-attn -> KT
  float* B4   = (float*)alloc(PAIR_B);   // bf16 transpose scratch / tri-attn VT
  float* QKV  = (float*)alloc((size_t)kL * kL * 3 * kTAH * kTAD * 4);   // 50.3 MB
  float* BPB  = (float*)alloc((size_t)kL * kL * kTAH * 4);
  float* LOG  = (float*)alloc((size_t)32 * kTAH * kL * kL * 4);         // 33.55 MB chunk
  __bf16* LOGB = (__bf16*)alloc((size_t)32 * kTAH * kL * kL * 2);       // bf16 probs

  // ---- pair state init ----
  hipMemcpyAsync(PAIR, d_in[IN_PAIR], PAIR_B, hipMemcpyDeviceToDevice, st);

  // ---- src path: mask, conv, norm3 ----
  k_maskmul<<<ewb(kL * kD), 256, 0, st>>>(src, mask, S0, kL * kD);
  k_conv<<<kL, kD, 0, st>>>(S0, PF(P_CONV_W), PF(P_CONV_B), S1);
  k_layernorm<<<kL, 32, 0, st>>>(S1, nullptr, S2, 0, PF(P_NORM3_G), PF(P_NORM3_B), kD, 1e-5f);

  // ---- pair -> MHA bias ----
  k_layernorm<<<kL * kL, 32, 0, st>>>(PAIR, nullptr, XNB, 1, PF(P_PWN_G), PF(P_PWN_B), kP, 1e-5f);
  { GemmDesc g = gd(XNB, PF(P_P2H_W), PWB, nullptr, kL * kL, kNH, kP); g.abf = 1; rungemm(st, g); }

  // ---- MHA ----
  { GemmDesc g = gd(S2, PF(P_MHA_WQ), S3, nullptr, kL, kD, kD); rungemm(st, g); }
  { GemmDesc g = gd(S2, PF(P_MHA_WK), S4, nullptr, kL, kD, kD); rungemm(st, g); }
  { GemmDesc g = gd(S2, PF(P_MHA_WV), S5, nullptr, kL, kD, kD); rungemm(st, g); }
  { // logits[h][i][j] = q_h . k_h / sqrt(32)   (fast path: K=32, contiguous heads)
    GemmDesc g = gd(S3, S4, ATT, nullptr, kL, kL, kDK);
    g.am = kD; g.abi = kDK;
    g.bn = kD; g.bbi = kDK;
    g.cm = kL; g.cn = 1; g.cbi = (long long)kL * kL;
    g.binner = kNH; g.alpha = 0.17677669529663687f;
    rungemm(st, g, kNH);
  }
  k_softmax<<<dim3(kL, kNH), 32, 0, st>>>(ATT, LOGB, PWB, kNH, mask, 0);
  // V -> bf16 [h][d][j] so AV is j-contiguous with no cvt.
  k_vtrans<<<ewb(kNH * kDK * kL), 256, 0, st>>>(S5, VTM, 0, kD, kDK, 0, 1, kDK, kNH);
  { // out[i, h*32+d] = attn_h @ v_h   (bf16/bf16 fast path)
    GemmDesc g = gd(LOGB, VTM, S6, nullptr, kL, kDK, kL);
    g.abi = (long long)kL * kL;
    g.bn = kL; g.bbi = (long long)kDK * kL;
    g.cm = kD; g.cn = 1; g.cbi = kDK;
    g.binner = kNH; g.abf = 1; g.bbf = 1;
    rungemm(st, g, kNH);
  }
  { GemmDesc g = gd(S6, PF(P_MHA_FC), S3, nullptr, kL, kD, kD); rungemm(st, g); }
  k_layernorm<<<kL, 32, 0, st>>>(S3, S2, S4, 0, PF(P_MHA_LN_G), PF(P_MHA_LN_B), kD, 1e-6f); // src2
  k_layernorm<<<kL, 32, 0, st>>>(S2, S4, S5, 0, PF(P_NORM1_G), PF(P_NORM1_B), kD, 1e-5f);   // src

  // ---- FFN ----
  { GemmDesc g = gd(S5, PF(P_LIN1_W), FFB, PF(P_LIN1_B), kL, kFF, kD); rungemm(st, g); }
  k_gelu<<<ewb(kL * kFF), 256, 0, st>>>(FFB, kL * kFF);
  { GemmDesc g = gd(FFB, PF(P_LIN2_W), S3, PF(P_LIN2_B), kL, kD, kFF); rungemm(st, g); }
  k_layernorm<<<kL, 32, 0, st>>>(S5, S3, S6, 0, PF(P_NORM2_G), PF(P_NORM2_B), kD, 1e-5f);   // final src

  // ---- outer product mean -> pair ----
  { GemmDesc g = gd(S6, PF(P_OPM1_W), SVEC, PF(P_OPM1_B), kL, kDM, kD); rungemm(st, g); }
  { // t[i, p*32+c] = sum_d s[i,d] * W2[p, c*32+d]
    GemmDesc g = gd(SVEC, PF(P_OPM2_W), TBUF, nullptr, kL, kP * kDM, kDM);
    rungemm(st, g);   // bn=K=32 matches the flat layout; fast path
  }
  { // pair[i,j,p] += sum_c s[j,c] * t[i,p,c] + opm2_b[p]   (batched over i; fast path)
    GemmDesc g = gd(SVEC, TBUF, PAIR, PF(P_OPM2_B), kL, kP, kDM);
    g.bbi = (long long)kP * kDM;
    g.cm = kP; g.cn = 1; g.cbi = (long long)kL * kP;
    g.binner = kL; g.accum = 1;
    rungemm(st, g, kL);
  }

  // ---- pair stack ----
  tri_mult_block(st, d_in, TMO_BASE, PAIR, XNB, B0, B1, B2, B3, B4, mask, true);
  tri_mult_block(st, d_in, TMI_BASE, PAIR, XNB, B0, B1, B2, B3, B4, mask, false);
  tri_attn_block(st, d_in, TAO_BASE, PAIR, XNB, B0, B1, QKV, BPB, LOG, LOGB,
                 (__bf16*)B2, (__bf16*)B3, (__bf16*)B4, mask, true);
  tri_attn_block(st, d_in, TAI_BASE, PAIR, XNB, B0, B1, QKV, BPB, LOG, LOGB,
                 (__bf16*)B2, (__bf16*)B3, (__bf16*)B4, mask, false);

  // ---- pair transition (chunked so the 65536x256 hidden fits one pair buffer) ----
  k_layernorm<<<kL * kL, 32, 0, st>>>(PAIR, nullptr, XNB, 1, PF(P_PT_LN_G), PF(P_PT_LN_B), kP, 1e-5f);
  for (int r0 = 0; r0 < kL * kL; r0 += 16384) {
    GemmDesc g1 = gd(XNB, PF(P_PT_W1), B0, PF(P_PT_B1), 16384, 4 * kP, kP);
    g1.abase = (long long)r0 * kP; g1.abf = 1;
    rungemm(st, g1);
    k_relu<<<ewb(16384 * 4 * kP), 256, 0, st>>>(B0, 16384 * 4 * kP);
    GemmDesc g2 = gd(B0, PF(P_PT_W2), PAIR, PF(P_PT_B2), 16384, kP, 4 * kP);
    g2.cbase = (long long)r0 * kP; g2.accum = 1;
    rungemm(st, g2);
  }

  // ---- outputs: (src, pair) concatenated flat ----
  hipMemcpyAsync(d_out, S6, SRC_B, hipMemcpyDeviceToDevice, st);
  hipMemcpyAsync((float*)d_out + (size_t)kL * kD, PAIR, PAIR_B, hipMemcpyDeviceToDevice, st);
}
